// DevignModel_5025111736768
// MI455X (gfx1250) — matrix operand order repaired and checked by reading, not compiled, not run on hardware
//
// Devign model (GGNN + conv readout) for MI455X / gfx1250.
// wave32 + v_wmma_f32_16x16x32_bf16 + async global->LDS staging (ASYNCcnt).
#include <hip/hip_runtime.h>
#include <hip/hip_bf16.h>

typedef __attribute__((ext_vector_type(16))) __bf16 v16bf;
typedef __attribute__((ext_vector_type(8)))  float  v8f;

#define IN_DIM   100
#define OUT_DIM  200
#define OUTP     224   // 200 padded to multiple of 32
#define GDIM     600
#define GP       608   // 600 padded (3*OUT gates)
#define CCD      300
#define CCP      320   // 300 padded to multiple of 32
#define NET      4
#define NSTEPS   8

// ---------------------------------------------------------------- utilities
__global__ void zero_f32_kernel(float* p, long long n) {
  long long i = (long long)blockIdx.x * 256 + threadIdx.x;
  if (i < n) p[i] = 0.f;
}

__global__ void f32_to_bf16_kernel(__bf16* dst, const float* src, long long n) {
  long long i = (long long)blockIdx.x * 256 + threadIdx.x;
  if (i < n) dst[i] = (__bf16)src[i];
}

// Pack B matrix for WMMA GEMM: dst[Kpad x Npad] (row-major, bf16), zero-padded.
// dst[kk, nn] = src[(nn*Ksrc + kk)*T + t]  (i.e. transpose of [N,K] weight, tap t)
__global__ void pack_bt_kernel(__bf16* dst, const float* src,
                               int Kpad, int Npad, int Ksrc, int Nsrc, int T, int t) {
  long long i = (long long)blockIdx.x * 256 + threadIdx.x;
  long long tot = (long long)Kpad * Npad;
  if (i >= tot) return;
  int kk = (int)(i / Npad), nn = (int)(i % Npad);
  float v = 0.f;
  if (kk < Ksrc && nn < Nsrc) v = src[((size_t)nn * Ksrc + kk) * T + t];
  dst[i] = (__bf16)v;
}

// h0 = [features | zeros]; write f32 h (N rows used) and bf16 h (NP rows, pad zero)
__global__ void init_h_kernel(const float* feat, float* h, __bf16* hb, int N, int NP) {
  long long i = (long long)blockIdx.x * 256 + threadIdx.x;
  long long tot = (long long)NP * OUTP;
  if (i >= tot) return;
  int n = (int)(i / OUTP), c = (int)(i % OUTP);
  float v = 0.f;
  if (n < N && c < IN_DIM) v = feat[(size_t)n * IN_DIM + c];
  if (n < N) h[(size_t)n * OUTP + c] = v;
  hb[i] = (__bf16)v;
}

// concat(h, x) -> bf16 [NP x CCP], pad rows/cols zero
__global__ void pack_cc_kernel(const __bf16* hb, const float* feat, __bf16* cc, int N, int NP) {
  long long i = (long long)blockIdx.x * 256 + threadIdx.x;
  long long tot = (long long)NP * CCP;
  if (i >= tot) return;
  int n = (int)(i / CCP), c = (int)(i % CCP);
  __bf16 v = (__bf16)0.f;
  if (n < N) {
    if (c < OUT_DIM)       v = hb[(size_t)n * OUTP + c];
    else if (c < CCD)      v = (__bf16)feat[(size_t)n * IN_DIM + (c - OUT_DIM)];
  }
  cc[i] = v;
}

// ---------------------------------------------------------------- WMMA GEMM
// C[M x Npad] = A[M x Kpad](bf16) * B[Kpad x Npad](bf16) (+bias); out f32 or bf16.
// Block: 256 thr = 8 waves. Block tile 32(M) x 128(N); each wave owns one 16-col
// strip and two 16x16 M-subtiles -> 2 WMMAs per K-step on a shared B fragment.
// Tiles staged with GLOBAL_LOAD_ASYNC_TO_LDS (ASYNCcnt) per cdna5 §15.18.3.
__global__ void __launch_bounds__(256)
gemm_bf16_wmma_kernel(const __bf16* __restrict__ A, int lda,
                      const __bf16* __restrict__ B, int ldb,
                      float* __restrict__ Cf, __bf16* __restrict__ Cb, int ldc,
                      const float* __restrict__ bias, int biasN,
                      int M, int Npad, int Kpad, int accumulate) {
  __shared__ __align__(16) __bf16 lA[32 * 32];    // 2 KB
  __shared__ __align__(16) __bf16 lB[32 * 128];   // 8 KB
  const int tid  = threadIdx.x;
  const int wid  = tid >> 5;
  const int lane = tid & 31;
  const int m0     = blockIdx.y * 32;
  const int n0blk  = blockIdx.x * 128;
  const int nw     = n0blk + (wid << 4);
  const bool active = (nw < Npad);

  // cooperative-load coordinates (A: 1 x b64 / thread, B: 2 x b128 / thread)
  const int ar = tid >> 3;                 // A row 0..31
  const int ac = (tid & 7) * 4;            // A col, 4 bf16
  const unsigned ldsA = (unsigned)(uintptr_t)&lA[ar * 32 + ac];

  v8f acc0 = {}, acc1 = {};
  for (int k0 = 0; k0 < Kpad; k0 += 32) {
    {  // A tile 32x32 -> LDS, async
      const __bf16* ga = &A[(size_t)(m0 + ar) * lda + k0 + ac];
      asm volatile("global_load_async_to_lds_b64 %0, %1, off"
                   :: "v"(ldsA), "v"(ga) : "memory");
    }
#pragma unroll
    for (int it = 0; it < 2; ++it) {       // B tile 32x128 -> LDS, async
      int flat = (tid + it * 256) * 8;
      int br = flat >> 7, bc = flat & 127;
      if (n0blk + bc < Npad) {
        const __bf16* gb = &B[(size_t)(k0 + br) * ldb + n0blk + bc];
        unsigned ldsB = (unsigned)(uintptr_t)&lB[flat];
        asm volatile("global_load_async_to_lds_b128 %0, %1, off"
                     :: "v"(ldsB), "v"(gb) : "memory");
      }
    }
    if (k0 + 32 < Kpad)                    // prefetch next K-step of B strip
      __builtin_prefetch(&B[(size_t)(k0 + 32) * ldb + n0blk + (tid & 15) * 8], 0, 1);
    asm volatile("s_wait_asynccnt 0x0" ::: "memory");
    __syncthreads();
    if (active) {
      union { v16bf v; unsigned int u[8]; } Af0, Af1, Bf;
      const int mrow = lane & 15;
      const int g    = lane >> 4;
#pragma unroll
      for (int i = 0; i < 8; ++i) {        // A frag: ISA 16-bit 16x32 layout
        int e  = 2 * i;
        int kk = (e & 7) + ((e & 8) ? 16 : 0) + (g << 3);
        Af0.u[i] = *(const unsigned int*)&lA[mrow * 32 + kk];
        Af1.u[i] = *(const unsigned int*)&lA[(16 + mrow) * 32 + kk];
      }
#pragma unroll
      for (int i = 0; i < 16; ++i) {       // B frag: N = lane%16, K = ISA 16x32 layout
        int kk = (i & 7) + ((i & 8) ? 16 : 0) + (g << 3);
        Bf.v[i] = lB[kk * 128 + (wid << 4) + mrow];
      }
      acc0 = __builtin_amdgcn_wmma_f32_16x16x32_bf16(
          false, Af0.v, false, Bf.v, (short)0, acc0, false, false);
      acc1 = __builtin_amdgcn_wmma_f32_16x16x32_bf16(
          false, Af1.v, false, Bf.v, (short)0, acc1, false, false);
    }
    __syncthreads();
  }
  if (active) {
    const int col = nw + (lane & 15);
    const float bv = (bias && col < biasN) ? bias[col] : 0.f;
    const int g = lane >> 4;
#pragma unroll
    for (int v = 0; v < 8; ++v) {          // D layout: M = v + 8*(lane/16)
      int row = m0 + v + (g << 3);
      size_t i0 = (size_t)row * ldc + col;
      size_t i1 = (size_t)(row + 16) * ldc + col;
      float v0 = acc0[v] + bv, v1 = acc1[v] + bv;
      if (Cf) {
        if (accumulate) { Cf[i0] += v0; Cf[i1] += v1; }
        else            { Cf[i0]  = v0; Cf[i1]  = v1; }
      } else {
        Cb[i0] = (__bf16)v0; Cb[i1] = (__bf16)v1;
      }
    }
  }
}

// ------------------------------------------------------- edge gather/scatter
// a[dst] += t[etype, src]   (bf16 gather, f32 L2-resident atomics). 4 cols/thread.
__global__ void edge_scatter_kernel(const __bf16* __restrict__ t,
                                    const int* __restrict__ ei,
                                    const int* __restrict__ et,
                                    float* __restrict__ a, int E, int MN) {
  long long gid = (long long)blockIdx.x * 256 + threadIdx.x;
  if (gid >= (long long)E * 50) return;
  int e = (int)(gid / 50);
  int c = (int)(gid % 50) * 4;
  int src = ei[e], dst = ei[E + e], k = et[e];
  const __bf16* row = t + ((size_t)k * MN + src) * OUTP + c;
  uint2 p = *(const uint2*)row;
  const __bf16* pb = (const __bf16*)&p;
  float* ad = a + (size_t)dst * OUTP + c;
  atomicAdd(ad + 0, (float)pb[0]);
  atomicAdd(ad + 1, (float)pb[1]);
  atomicAdd(ad + 2, (float)pb[2]);
  atomicAdd(ad + 3, (float)pb[3]);
}

// ---------------------------------------------------------------- GRU update
__global__ void gru_update_kernel(const __bf16* __restrict__ gi,
                                  const __bf16* __restrict__ gh,
                                  float* __restrict__ h, __bf16* __restrict__ hb,
                                  int N) {
  long long i = (long long)blockIdx.x * 256 + threadIdx.x;
  if (i >= (long long)N * OUT_DIM) return;
  int n = (int)(i / OUT_DIM), c = (int)(i % OUT_DIM);
  size_t gb = (size_t)n * GP + c;
  float ir = (float)gi[gb], iz = (float)gi[gb + 200], in = (float)gi[gb + 400];
  float hr = (float)gh[gb], hz = (float)gh[gb + 200], hn = (float)gh[gb + 400];
  float r  = 1.f / (1.f + __expf(-(ir + hr)));
  float z  = 1.f / (1.f + __expf(-(iz + hz)));
  float nn = tanhf(in + r * hn);
  size_t hi = (size_t)n * OUTP + c;
  float hv = h[hi];
  float out = (1.f - z) * nn + z * hv;
  h[hi]  = out;
  hb[hi] = (__bf16)out;
}

// ------------------------------------------------------------- BN / pooling
__global__ void bn_stats_kernel(const float* __restrict__ x, int ld, int L,
                                float* mean, float* rstd) {
  int c = blockIdx.x, tid = threadIdx.x;
  float s = 0.f, s2 = 0.f;
  for (int l = tid; l < L; l += 256) {
    float v = x[(size_t)l * ld + c];
    s += v; s2 += v * v;
  }
  __shared__ float sh[256], sh2[256];
  sh[tid] = s; sh2[tid] = s2; __syncthreads();
  for (int off = 128; off > 0; off >>= 1) {
    if (tid < off) { sh[tid] += sh[tid + off]; sh2[tid] += sh2[tid + off]; }
    __syncthreads();
  }
  if (tid == 0) {
    float m = sh[0] / (float)L;
    float var = sh2[0] / (float)L - m * m;      // biased variance (jnp.var)
    mean[c] = m;
    rstd[c] = rsqrtf(var + 1e-5f);
  }
}

// fused BN-apply + ReLU + maxpool(k,s); out bf16 (next GEMM A) or f32 (final MLP)
__global__ void bn_relu_pool_kernel(const float* __restrict__ x, int ld, int C, int Cpad,
                                    const float* mean, const float* rstd,
                                    const float* g, const float* b,
                                    int k, int s, int Lvalid, int Lout,
                                    __bf16* outb, float* outf, int oldc) {
  long long i = (long long)blockIdx.x * 256 + threadIdx.x;
  if (i >= (long long)Lout * Cpad) return;
  int lp = (int)(i / Cpad), c = (int)(i % Cpad);
  float r = 0.f;
  if (lp < Lvalid && c < C) {
    float m = mean[c], rs = rstd[c], gg = g[c], bb = b[c];
    float best = 0.f;   // ReLU output >= 0
    for (int j = 0; j < k; ++j) {
      float v = (x[(size_t)(lp * s + j) * ld + c] - m) * rs * gg + bb;
      v = v > 0.f ? v : 0.f;
      best = v > best ? v : best;
    }
    r = best;
  }
  size_t o = (size_t)lp * oldc + c;
  if (outb) outb[o] = (__bf16)r; else outf[o] = r;
}

// ------------------------------------------------------------------- finale
__global__ void final_reduce_kernel(const float* __restrict__ Y2,
                                    const float* __restrict__ Z2,
                                    const float* myw, const float* myb,
                                    const float* mzw, const float* mzb,
                                    int L2, float* acc) {
  int l = blockIdx.x * 256 + threadIdx.x;
  float p0 = 0.f, p1 = 0.f;
  if (l < L2) {
    const float* yr = Y2 + (size_t)l * OUTP;
    float y0 = myb[0], y1 = myb[1];
    for (int c = 0; c < OUT_DIM; ++c) { float v = yr[c]; y0 += v * myw[c]; y1 += v * myw[OUT_DIM + c]; }
    const float* zr = Z2 + (size_t)l * CCP;
    float z0 = mzb[0], z1 = mzb[1];
    for (int c = 0; c < CCD; ++c) { float v = zr[c]; z0 += v * mzw[c]; z1 += v * mzw[CCD + c]; }
    p0 = y0 * z0; p1 = y1 * z1;
  }
  __shared__ float s0[256], s1[256];
  s0[threadIdx.x] = p0; s1[threadIdx.x] = p1; __syncthreads();
  for (int off = 128; off > 0; off >>= 1) {
    if (threadIdx.x < off) { s0[threadIdx.x] += s0[threadIdx.x + off];
                             s1[threadIdx.x] += s1[threadIdx.x + off]; }
    __syncthreads();
  }
  if (threadIdx.x == 0) { atomicAdd(&acc[0], s0[0]); atomicAdd(&acc[1], s1[0]); }
}

__global__ void final_out_kernel(const float* acc, float invL, float* out) {
  int i = threadIdx.x;
  if (i < 2) out[i] = 1.f / (1.f + __expf(-acc[i] * invL));
}

// ================================================================== launch
extern "C" void kernel_launch(void* const* d_in, const int* in_sizes, int n_in,
                              void* d_out, int out_size, void* d_ws, size_t ws_size,
                              hipStream_t stream) {
  (void)n_in; (void)out_size; (void)ws_size;
  const float* features = (const float*)d_in[0];
  const int*   eidx     = (const int*)  d_in[1];
  const int*   etyp     = (const int*)  d_in[2];
  const float* ggnn_W   = (const float*)d_in[3];
  const float* ggnn_b   = (const float*)d_in[4];
  const float* gru_Wih  = (const float*)d_in[5];
  const float* gru_Whh  = (const float*)d_in[6];
  const float* gru_bih  = (const float*)d_in[7];
  const float* gru_bhh  = (const float*)d_in[8];
  const float* conv1_w  = (const float*)d_in[9];
  const float* conv1_b  = (const float*)d_in[10];
  const float* conv2_w  = (const float*)d_in[11];
  const float* conv2_b  = (const float*)d_in[12];
  const float* convc1_w = (const float*)d_in[13];
  const float* convc1_b = (const float*)d_in[14];
  const float* convc2_w = (const float*)d_in[15];
  const float* convc2_b = (const float*)d_in[16];
  const float* bn_y_g   = (const float*)d_in[17];
  const float* bn_y_b   = (const float*)d_in[18];
  const float* bn_c_g   = (const float*)d_in[19];
  const float* bn_c_b   = (const float*)d_in[20];
  const float* mlp_y_w  = (const float*)d_in[21];
  const float* mlp_y_b  = (const float*)d_in[22];
  const float* mlp_z_w  = (const float*)d_in[23];
  const float* mlp_z_b  = (const float*)d_in[24];

  const int N  = in_sizes[0] / IN_DIM;       // 100000
  const int E  = in_sizes[2];                // 1600000
  const int MN = ((N + 31) / 32) * 32;       // GEMM row count (32-tiled M)
  const int NP = MN + 16;                    // + slack rows for conv tap shifts

  // ------------------------------------------------------- workspace carve
  char* base = (char*)d_ws;
  size_t off = 0;
  auto carve = [&](size_t bytes) -> char* {
    char* p = base + off; off = (off + bytes + 255) & ~(size_t)255; return p;
  };
  __bf16* wWt  = (__bf16*)carve((size_t)NET * OUTP * OUTP * 2);
  __bf16* wWih = (__bf16*)carve((size_t)OUTP * GP * 2);
  __bf16* wWhh = (__bf16*)carve((size_t)OUTP * GP * 2);
  __bf16* wC1  = (__bf16*)carve((size_t)3 * OUTP * OUTP * 2);
  __bf16* wC2  = (__bf16*)carve((size_t)OUTP * OUTP * 2);
  __bf16* wZ1  = (__bf16*)carve((size_t)3 * CCP * CCP * 2);
  __bf16* wZ2  = (__bf16*)carve((size_t)CCP * CCP * 2);
  float*  statM = (float*)carve(CCP * 4);
  float*  statR = (float*)carve(CCP * 4);
  float*  accP  = (float*)carve(2 * 4);
  __bf16* hb   = (__bf16*)carve((size_t)NP * OUTP * 2);
  __bf16* ccb  = (__bf16*)carve((size_t)NP * CCP * 2);

  char* arena = base + off;                  // GGNN scratch (dead after steps)
  size_t ao = 0;
  auto acarve = [&](size_t bytes) -> char* {
    char* p = arena + ao; ao = (ao + bytes + 255) & ~(size_t)255; return p;
  };
  float*  hF  = (float*) acarve((size_t)MN * OUTP * 4);
  float*  aF  = (float*) acarve((size_t)MN * OUTP * 4);
  __bf16* abB = (__bf16*)acarve((size_t)MN * OUTP * 2);
  __bf16* tB  = (__bf16*)acarve((size_t)NET * MN * OUTP * 2);
  __bf16* giB = (__bf16*)acarve((size_t)MN * GP * 2);
  __bf16* ghB = (__bf16*)acarve((size_t)MN * GP * 2);

  const int L1  = N - 2;                     // conv1/convc1 valid length
  const int M1  = ((L1 + 31) / 32) * 32;
  const int LpY = (L1 - 3) / 2 + 1;          // after maxpool(3,2)
  const int M2  = ((LpY + 31) / 32) * 32;
  const int L2  = (LpY - 2) / 2 + 1;         // after maxpool(2,2)

  size_t ro = 0;                             // readout views alias the arena
  auto rcarve = [&](size_t bytes) -> char* {
    char* p = arena + ro; ro = (ro + bytes + 255) & ~(size_t)255; return p;
  };
  float*  yfull = (float*) rcarve((size_t)M1 * OUTP * 4);
  float*  zfull = (float*) rcarve((size_t)M1 * CCP * 4);
  __bf16* poolY = (__bf16*)rcarve((size_t)M2 * OUTP * 2);
  __bf16* poolZ = (__bf16*)rcarve((size_t)M2 * CCP * 2);
  float*  y2f   = (float*) rcarve((size_t)M2 * OUTP * 4);
  float*  z2f   = (float*) rcarve((size_t)M2 * CCP * 4);
  float*  Y2f   = (float*) rcarve((size_t)L2 * OUTP * 4);
  float*  Z2f   = (float*) rcarve((size_t)L2 * CCP * 4);

  auto blk1d = [](long long n) -> dim3 { return dim3((unsigned)((n + 255) / 256)); };
  auto gemm = [&](const __bf16* A, int lda, const __bf16* B, int ldb,
                  float* Cf, __bf16* Cb, int ldc, const float* bias, int biasN,
                  int M, int Npad, int Kpad, int accum) {
    dim3 g((Npad + 127) / 128, M / 32);
    gemm_bf16_wmma_kernel<<<g, 256, 0, stream>>>(A, lda, B, ldb, Cf, Cb, ldc,
                                                 bias, biasN, M, Npad, Kpad, accum);
  };

  // ----------------------------------------------------------- pack weights
  for (int k = 0; k < NET; ++k)
    pack_bt_kernel<<<blk1d((long long)OUTP * OUTP), 256, 0, stream>>>(
        wWt + (size_t)k * OUTP * OUTP, ggnn_W + (size_t)k * OUT_DIM * OUT_DIM,
        OUTP, OUTP, OUT_DIM, OUT_DIM, 1, 0);
  pack_bt_kernel<<<blk1d((long long)OUTP * GP), 256, 0, stream>>>(
      wWih, gru_Wih, OUTP, GP, OUT_DIM, GDIM, 1, 0);
  pack_bt_kernel<<<blk1d((long long)OUTP * GP), 256, 0, stream>>>(
      wWhh, gru_Whh, OUTP, GP, OUT_DIM, GDIM, 1, 0);
  for (int t = 0; t < 3; ++t)
    pack_bt_kernel<<<blk1d((long long)OUTP * OUTP), 256, 0, stream>>>(
        wC1 + (size_t)t * OUTP * OUTP, conv1_w, OUTP, OUTP, OUT_DIM, OUT_DIM, 3, t);
  pack_bt_kernel<<<blk1d((long long)OUTP * OUTP), 256, 0, stream>>>(
      wC2, conv2_w, OUTP, OUTP, OUT_DIM, OUT_DIM, 1, 0);
  for (int t = 0; t < 3; ++t)
    pack_bt_kernel<<<blk1d((long long)CCP * CCP), 256, 0, stream>>>(
        wZ1 + (size_t)t * CCP * CCP, convc1_w, CCP, CCP, CCD, CCD, 3, t);
  pack_bt_kernel<<<blk1d((long long)CCP * CCP), 256, 0, stream>>>(
      wZ2, convc2_w, CCP, CCP, CCD, CCD, 1, 0);

  // -------------------------------------------------------------- GGNN loop
  init_h_kernel<<<blk1d((long long)NP * OUTP), 256, 0, stream>>>(features, hF, hb, N, NP);
  for (int s = 0; s < NSTEPS; ++s) {
    zero_f32_kernel<<<blk1d((long long)MN * OUTP), 256, 0, stream>>>(aF, (long long)MN * OUTP);
    for (int k = 0; k < NET; ++k)                       // t_k = h @ W_k^T + b_k
      gemm(hb, OUTP, wWt + (size_t)k * OUTP * OUTP, OUTP,
           nullptr, tB + (size_t)k * MN * OUTP, OUTP,
           ggnn_b + (size_t)k * OUT_DIM, OUT_DIM, MN, OUTP, OUTP, 0);
    edge_scatter_kernel<<<blk1d((long long)E * 50), 256, 0, stream>>>(tB, eidx, etyp, aF, E, MN);
    f32_to_bf16_kernel<<<blk1d((long long)MN * OUTP), 256, 0, stream>>>(abB, aF, (long long)MN * OUTP);
    gemm(abB, OUTP, wWih, GP, nullptr, giB, GP, gru_bih, GDIM, MN, GP, OUTP, 0);
    gemm(hb,  OUTP, wWhh, GP, nullptr, ghB, GP, gru_bhh, GDIM, MN, GP, OUTP, 0);
    gru_update_kernel<<<blk1d((long long)N * OUT_DIM), 256, 0, stream>>>(giB, ghB, hF, hb, N);
  }

  // ---------------------------------------------------------------- readout
  pack_cc_kernel<<<blk1d((long long)NP * CCP), 256, 0, stream>>>(hb, features, ccb, N, NP);

  // Y branch: conv1(k=3) = 3 row-shifted GEMMs -> BN -> relu+pool3 -> conv2 -> BN -> relu+pool2
  for (int t = 0; t < 3; ++t)
    gemm(hb + (size_t)t * OUTP, OUTP, wC1 + (size_t)t * OUTP * OUTP, OUTP,
         yfull, nullptr, OUTP, t == 0 ? conv1_b : nullptr, OUT_DIM, M1, OUTP, OUTP, t > 0);
  bn_stats_kernel<<<OUT_DIM, 256, 0, stream>>>(yfull, OUTP, L1, statM, statR);
  bn_relu_pool_kernel<<<blk1d((long long)M2 * OUTP), 256, 0, stream>>>(
      yfull, OUTP, OUT_DIM, OUTP, statM, statR, bn_y_g, bn_y_b, 3, 2, LpY, M2, poolY, nullptr, OUTP);
  gemm(poolY, OUTP, wC2, OUTP, y2f, nullptr, OUTP, conv2_b, OUT_DIM, M2, OUTP, OUTP, 0);
  bn_stats_kernel<<<OUT_DIM, 256, 0, stream>>>(y2f, OUTP, LpY, statM, statR);
  bn_relu_pool_kernel<<<blk1d((long long)L2 * OUTP), 256, 0, stream>>>(
      y2f, OUTP, OUT_DIM, OUTP, statM, statR, bn_y_g, bn_y_b, 2, 2, L2, L2, nullptr, Y2f, OUTP);

  // Z branch (concat features)
  for (int t = 0; t < 3; ++t)
    gemm(ccb + (size_t)t * CCP, CCP, wZ1 + (size_t)t * CCP * CCP, CCP,
         zfull, nullptr, CCP, t == 0 ? convc1_b : nullptr, CCD, M1, CCP, CCP, t > 0);
  bn_stats_kernel<<<CCD, 256, 0, stream>>>(zfull, CCP, L1, statM, statR);
  bn_relu_pool_kernel<<<blk1d((long long)M2 * CCP), 256, 0, stream>>>(
      zfull, CCP, CCD, CCP, statM, statR, bn_c_g, bn_c_b, 3, 2, LpY, M2, poolZ, nullptr, CCP);
  gemm(poolZ, CCP, wZ2, CCP, z2f, nullptr, CCP, convc2_b, CCD, M2, CCP, CCP, 0);
  bn_stats_kernel<<<CCD, 256, 0, stream>>>(z2f, CCP, LpY, statM, statR);
  bn_relu_pool_kernel<<<blk1d((long long)L2 * CCP), 256, 0, stream>>>(
      z2f, CCP, CCD, CCP, statM, statR, bn_c_g, bn_c_b, 2, 2, L2, L2, nullptr, Z2f, CCP);

  // final: mean over L of (Y2@Wy + by) * (Z2@Wz + bz), then sigmoid
  zero_f32_kernel<<<1, 256, 0, stream>>>(accP, 2);
  final_reduce_kernel<<<blk1d(L2), 256, 0, stream>>>(Y2f, Z2f, mlp_y_w, mlp_y_b,
                                                     mlp_z_w, mlp_z_b, L2, accP);
  final_out_kernel<<<1, 32, 0, stream>>>(accP, 1.0f / (float)L2, (float*)d_out);
}
